// FlattenedMaskedLocalAttention2d_46110768890407
// MI455X (gfx1250) — matrix operand order repaired and checked by reading, not compiled, not run on hardware
//
#include <hip/hip_runtime.h>
#include <hip/hip_bf16.h>

typedef __attribute__((ext_vector_type(16))) _Float16 v16h;
typedef __attribute__((ext_vector_type(8)))  float    v8f;

#define HD   16      // head dim
#define HW   64      // height == width
#define NTOK 4096    // HW*HW
#define RAD  3       // window radius (7//2), dilation 1

// One wave32 handles one tile of 16 queries (one row segment) for one (b, head).
__global__ __launch_bounds__(128) void natten2d_wmma_kernel(
    const float* __restrict__ q, const float* __restrict__ k,
    const float* __restrict__ v, float* __restrict__ out)
{
    const int  lane = threadIdx.x;        // 0..31
    const bool hi   = lane >= 16;
    const int  nlo  = lane & 15;

    const int waveId = blockIdx.x * blockDim.y + threadIdx.y; // 0..2047
    const int b    = waveId >> 10;        // 2 batches * 4 heads * 256 tiles
    const int h    = (waveId >> 8) & 3;
    const int tile = waveId & 255;
    const int qr   = tile >> 2;           // query row
    const int qc0  = (tile & 3) << 4;     // first query column of tile

    const size_t chanBase = (size_t)(b * 64 + h * HD) * NTOK;

    // ---------- B operand of S^T = K_tile @ Q^T : Q^T is 32(dpad) x 16(queries) ----------
    // B layout: elems 0-7 -> d = e (lanes 0-15) / d = 8+e (lanes 16-31); elems 8-15 (d=16..31) are pad=0.
    v16h b_q;
#pragma unroll
    for (int e = 0; e < 16; ++e) b_q[e] = (_Float16)0.f;
    {
        const float* qb = q + chanBase + (size_t)qr * HW + (qc0 + nlo);
#pragma unroll
        for (int e = 0; e < 8; ++e)
            b_q[e] = (_Float16)(qb[(size_t)(e + (hi ? 8 : 0)) * NTOK] * 0.25f);  // fold 1/sqrt(16)
    }

    // Flash state: per-lane stats are for query n = lane%16 (replicated in lane pair).
    float run_max = -1e30f;
    float run_sum = 0.f;
    v8f o;                                  // O^T accumulator: lane = query, VGPR j = d (j or j+8)
#pragma unroll
    for (int j = 0; j < 8; ++j) o[j] = 0.f;

    const int kc0 = qc0 - 8;                // 32-key strip covers all columns within radius
    const int qc  = qc0 + nlo;

    for (int dr = -RAD; dr <= RAD; ++dr) {
        const int kr = qr + dr;
        if (kr < 0 || kr >= HW) continue;   // wave-uniform branch: EXEC stays full
        const size_t rowBase = chanBase + (size_t)kr * HW;

        // ---- A operands: K_tile chunks, 16(keys) x 32(dpad) ----
        // A layout: m = lane%16; lane<16: elems 0-7 -> d 0..7 ; lane>=16: elems 0-7 -> d 8..15;
        // elems 8-15 map to d 16..31 -> zero pad.
        v16h a_k0, a_k1;
#pragma unroll
        for (int e = 0; e < 16; ++e) { a_k0[e] = (_Float16)0.f; a_k1[e] = (_Float16)0.f; }
        {
            const int kcA = min(max(kc0 + nlo,      0), HW - 1);
            const int kcB = min(max(kc0 + 16 + nlo, 0), HW - 1);
            const float* kbA = k + rowBase + kcA;
            const float* kbB = k + rowBase + kcB;
#pragma unroll
            for (int j = 0; j < 8; ++j) {
                const int d = j + (hi ? 8 : 0);
                a_k0[j] = (_Float16)kbA[(size_t)d * NTOK];
                a_k1[j] = (_Float16)kbB[(size_t)d * NTOK];
            }
        }

        v8f zacc;
#pragma unroll
        for (int j = 0; j < 8; ++j) zacc[j] = 0.f;

        // S^T tiles: D[key][query]; lane = query, VGPR j = key j (lanes<16) / key j+8 (lanes>=16)
        v8f s0 = __builtin_amdgcn_wmma_f32_16x16x32_f16(false, a_k0, false, b_q,
                                                        (short)0, zacc, false, false);
        v8f s1 = __builtin_amdgcn_wmma_f32_16x16x32_f16(false, a_k1, false, b_q,
                                                        (short)0, zacc, false, false);

        // ---- mask + online softmax stats (per query = per lane pair) ----
        float p0[8], p1[8];
        float tmax = -1e30f;
#pragma unroll
        for (int j = 0; j < 8; ++j) {
            const int kk0 = j + (hi ? 8 : 0);
            const int kcA = kc0 + kk0;
            const int kcB = kcA + 16;
            const int dA  = qc - kcA, dB = qc - kcB;
            const bool okA = (kcA >= 0) && (kcA < HW) && (dA <= RAD) && (dA >= -RAD);
            const bool okB = (kcB >= 0) && (kcB < HW) && (dB <= RAD) && (dB >= -RAD);
            p0[j] = okA ? s0[j] : -1e30f;
            p1[j] = okB ? s1[j] : -1e30f;
            tmax = fmaxf(tmax, fmaxf(p0[j], p1[j]));
        }
        tmax = fmaxf(tmax, __shfl_xor(tmax, 16, 32));   // lane pair -> full 32-key max
        const float mnew  = fmaxf(run_max, tmax);
        const float alpha = __expf(run_max - mnew);
        float psum = 0.f;
#pragma unroll
        for (int j = 0; j < 8; ++j) {
            p0[j] = __expf(p0[j] - mnew);
            p1[j] = __expf(p1[j] - mnew);
            psum += p0[j] + p1[j];
        }
        psum += __shfl_xor(psum, 16, 32);
        run_sum = run_sum * alpha + psum;
        run_max = mnew;

        // ---- build B = P^T (32 keys x 16 queries) straight from S^T C/D layout ----
        // lane<16 needs keys 0..15: keys 0-7 local (p0), keys 8-15 from partner's p0.
        // lane>=16 needs keys 16..31: keys 16-23 from partner's p1, keys 24-31 local (p1).
        v16h b_p;
#pragma unroll
        for (int j = 0; j < 8; ++j) {
            const float x0 = __shfl_xor(p0[j], 16, 32);
            const float x1 = __shfl_xor(p1[j], 16, 32);
            if (!hi) { b_p[j] = (_Float16)p0[j]; b_p[j + 8] = (_Float16)x0; }
            else     { b_p[j] = (_Float16)x1;    b_p[j + 8] = (_Float16)p1[j]; }
        }

        // ---- A = V^T (16 d x 32 keys), masked keys have p=0 so clamped loads are safe ----
        v16h a_v;
        const float* vb = v + chanBase + (size_t)nlo * NTOK + (size_t)kr * HW;
#pragma unroll
        for (int e = 0; e < 16; ++e) {
            const int kk = e + (hi ? 16 : 0);
            const int kc = min(max(kc0 + kk, 0), HW - 1);
            a_v[e] = (_Float16)vb[kc];
        }

        // ---- O^T = O^T*alpha + V^T @ P^T ----
#pragma unroll
        for (int j = 0; j < 8; ++j) o[j] *= alpha;
        o = __builtin_amdgcn_wmma_f32_16x16x32_f16(false, a_v, false, b_p,
                                                   (short)0, o, false, false);
    }

    // ---- normalize and write: out[(b,h*16+d), qr, qc0+n] ----
    const float inv = 1.f / run_sum;
    float* ob = out + chanBase + (size_t)qr * HW + (qc0 + nlo);
#pragma unroll
    for (int j = 0; j < 8; ++j) {
        const int d = j + (hi ? 8 : 0);
        ob[(size_t)d * NTOK] = o[j] * inv;
    }
}

extern "C" void kernel_launch(void* const* d_in, const int* in_sizes, int n_in,
                              void* d_out, int out_size, void* d_ws, size_t ws_size,
                              hipStream_t stream) {
    (void)in_sizes; (void)n_in; (void)out_size; (void)d_ws; (void)ws_size;
    const float* q = (const float*)d_in[0];
    const float* k = (const float*)d_in[1];
    const float* v = (const float*)d_in[2];
    float* out = (float*)d_out;

    // 2 batches * 4 heads * 256 query tiles = 2048 waves; 4 waves (128 threads) per block.
    dim3 block(32, 4, 1);
    dim3 grid(512, 1, 1);
    natten2d_wmma_kernel<<<grid, block, 0, stream>>>(q, k, v, out);
}
